// MoleculeVAE_36472862278406
// MI455X (gfx1250) — matrix operand order repaired and checked by reading, not compiled, not run on hardware
//
#include <hip/hip_runtime.h>
#include <hip/hip_bf16.h>
#include <stdint.h>

#define VOCAB 64
#define EMB   512
#define HID   1024
#define LAT   256
#define MAXLEN 100
#define BATCH 512
#define H3    (3*HID)

typedef __attribute__((ext_vector_type(16))) __bf16 v16bf;
typedef __attribute__((ext_vector_type(8)))  float  v8f;
typedef __attribute__((ext_vector_type(4)))  int    v4i;

union Frag { v16bf v; uint4 q[2]; };

// ---- optional CDNA5 async global->LDS path (probe-guarded) -----------------
#if defined(__has_builtin)
#  if __has_builtin(__builtin_amdgcn_global_load_async_to_lds_b128)
#    define HAVE_ASYNC_LDS 1
#  endif
#endif
#ifndef HAVE_ASYNC_LDS
#  define HAVE_ASYNC_LDS 0
#endif

#if HAVE_ASYNC_LDS
__device__ __forceinline__ void async_ld128(const void* gsrc, void* ldst) {
  __builtin_amdgcn_global_load_async_to_lds_b128(
      (__attribute__((address_space(1))) v4i*)gsrc,
      (__attribute__((address_space(3))) v4i*)ldst, 0, 0);
}
__device__ __forceinline__ void wait_async0() {
#  if __has_builtin(__builtin_amdgcn_s_wait_asynccnt)
  __builtin_amdgcn_s_wait_asynccnt(0);
#  else
  asm volatile("s_wait_asynccnt 0x0" ::: "memory");
#  endif
}
#endif

__device__ __forceinline__ unsigned short f2bfu(float f) {
  union { float f; uint32_t u; } a; a.f = f;
  uint32_t r = a.u + 0x7fffu + ((a.u >> 16) & 1u);   // RNE
  return (unsigned short)(r >> 16);
}
__device__ __forceinline__ float sigf(float x) {
  return __builtin_amdgcn_rcpf(1.0f + __expf(-x));
}
__device__ __forceinline__ float tanh_fast(float x) {
  return 1.0f - 2.0f * __builtin_amdgcn_rcpf(__expf(2.0f * x) + 1.0f);
}

// ---------------------------------------------------------------------------
// Precompute gate table: tab[v, j] = sum_e emb[v,e]*Wih[j,e] + bih[j]
// ---------------------------------------------------------------------------
__global__ void k_table(const float* __restrict__ emb, const float* __restrict__ Wih,
                        const float* __restrict__ bih, float* __restrict__ tab) {
  __shared__ __align__(16) float es[EMB];
  int v = blockIdx.x;
  int j = blockIdx.y * 256 + threadIdx.x;
  es[threadIdx.x]       = emb[v * EMB + threadIdx.x];
  es[threadIdx.x + 256] = emb[v * EMB + threadIdx.x + 256];
  __syncthreads();
  const float4* w4 = (const float4*)(Wih + (size_t)j * EMB);
  const float4* e4 = (const float4*)es;
  float acc = 0.f;
  for (int k = 0; k < EMB / 4; ++k) {
    float4 a = e4[k], b = w4[k];
    acc += a.x*b.x + a.y*b.y + a.z*b.z + a.w*b.w;
  }
  tab[(size_t)v * H3 + j] = acc + bih[j];
}

__global__ void k_cvt_bf16(const float* __restrict__ src, unsigned short* __restrict__ dst, int n) {
  for (int i = blockIdx.x * blockDim.x + threadIdx.x; i < n; i += gridDim.x * blockDim.x)
    dst[i] = f2bfu(src[i]);
}

__global__ void k_zero_h(float* __restrict__ hf, unsigned short* __restrict__ hb, int n) {
  for (int i = blockIdx.x * blockDim.x + threadIdx.x; i < n; i += gridDim.x * blockDim.x) {
    hf[i] = 0.f; hb[i] = 0;
  }
}

// ---------------------------------------------------------------------------
// Fused GRU step:  gh = h @ Whh^T (bf16 WMMA, f32 acc) for all 3 gates, then
// the gate nonlinearities + h update. Block tile 64 (B) x 32 (H), 8 waves.
// LDS double-buffered; staging is straight-line with hoisted addresses;
// GLOBAL_LOAD_ASYNC_TO_LDS_B128 when the toolchain exposes it. grid (8, 32).
// ---------------------------------------------------------------------------
__global__ void k_gru_step(const unsigned short* __restrict__ hA,   // prev h, bf16 [B,H]
                           const float* __restrict__ hF,            // prev h, f32  [B,H]
                           const unsigned short* __restrict__ Whh,  // bf16 [3H,H]
                           const float* __restrict__ bhh,           // [3H]
                           const float* __restrict__ gitab,         // [64,3H] (bih baked in)
                           const int* __restrict__ x, int t, int dec_mode,
                           float* __restrict__ hFo, unsigned short* __restrict__ hAo,
                           unsigned short* __restrict__ hall)       // bf16 [B,H] slice or null
{
  // row stride 40 shorts = 80B = 5*16B : 16B-aligned chunks, non-pow2 banking
  __shared__ __align__(16) unsigned short As[2][64][40];      // [buf][row][k]
  __shared__ __align__(16) unsigned short Bs[2][3][32][40];   // [buf][gate][n][k]

  const int tid  = threadIdx.x;
  const int lane = tid & 31;
  const int wave = tid >> 5;
  const int wm   = wave & 3;   // M subtile
  const int wn   = wave >> 2;  // N subtile
  const int b0   = blockIdx.x * 64;
  const int n0   = blockIdx.y * 32;
  const int half = lane >> 4;
  const int ml   = lane & 15;
  const bool extra = (tid < 128);

  // ---- loop-invariant staging addresses (advance 4 uint4 = 32 shorts / step)
  const int rowA = tid >> 2, chA = tid & 3;
  const uint4* gA = (const uint4*)(hA + (size_t)(b0 + rowA) * HID + chA * 8);
  const int g0 = tid >> 7, r0 = tid & 127, n0r = r0 >> 2, c0 = r0 & 3;
  const uint4* gB0 = (const uint4*)(Whh + (size_t)(g0 * HID + n0 + n0r) * HID + c0 * 8);
  const int i1 = tid + 256, g1 = i1 >> 7, r1 = i1 & 127, n1r = r1 >> 2, c1 = r1 & 3;
  const uint4* gB1 = (const uint4*)(Whh + (size_t)(g1 * HID + n0 + n1r) * HID + c1 * 8);
  uint4* dA[2]  = { (uint4*)&As[0][rowA][chA * 8],      (uint4*)&As[1][rowA][chA * 8] };
  uint4* dB0[2] = { (uint4*)&Bs[0][g0][n0r][c0 * 8],    (uint4*)&Bs[1][g0][n0r][c0 * 8] };
  uint4* dB1[2] = { (uint4*)&Bs[0][g1][n1r][c1 * 8],    (uint4*)&Bs[1][g1][n1r][c1 * 8] };

  v8f accR = {0,0,0,0,0,0,0,0};
  v8f accZ = {0,0,0,0,0,0,0,0};
  v8f accN = {0,0,0,0,0,0,0,0};

  const int NK = HID / 32;

  // ---- prologue: stage tile 0 into buffer 0
#if HAVE_ASYNC_LDS
  async_ld128(gA, dA[0]);
  async_ld128(gB0, dB0[0]);
  if (extra) async_ld128(gB1, dB1[0]);
  wait_async0();
#else
  {
    uint4 ra = *gA, rb0 = *gB0;
    *dA[0] = ra; *dB0[0] = rb0;
    if (extra) *dB1[0] = *gB1;
  }
#endif
  gA += 4; gB0 += 4; gB1 += 4;
  __syncthreads();

  for (int kt = 0; kt < NK; ++kt) {
    const int cur = kt & 1, nxt = cur ^ 1;
    const bool more = (kt + 1 < NK);

#if HAVE_ASYNC_LDS
    if (more) {
      async_ld128(gA, dA[nxt]);
      async_ld128(gB0, dB0[nxt]);
      if (extra) async_ld128(gB1, dB1[nxt]);
      gA += 4; gB0 += 4; gB1 += 4;
    }
#else
    uint4 ra, rb0, rb1;
    if (more) {
      ra  = *gA;
      rb0 = *gB0;
      if (extra) rb1 = *gB1;
      gA += 4; gB0 += 4; gB1 += 4;
    }
#endif

    // A fragment: two contiguous 8-elem K runs -> 2x ds_load_b128
    Frag fa;
    {
      const unsigned short* ar = &As[cur][wm * 16 + ml][0];
      fa.q[0] = *(const uint4*)(ar + half * 8);
      fa.q[1] = *(const uint4*)(ar + 16 + half * 8);
    }
#pragma unroll
    for (int g = 0; g < 3; ++g) {
      Frag fb;
      const unsigned short* br = &Bs[cur][g][wn * 16 + ml][0];
      fb.q[0] = *(const uint4*)(br + half * 8);
      fb.q[1] = *(const uint4*)(br + 16 + half * 8);
      if (g == 0)
        accR = __builtin_amdgcn_wmma_f32_16x16x32_bf16(false, fa.v, false, fb.v, (short)0, accR, false, false);
      else if (g == 1)
        accZ = __builtin_amdgcn_wmma_f32_16x16x32_bf16(false, fa.v, false, fb.v, (short)0, accZ, false, false);
      else
        accN = __builtin_amdgcn_wmma_f32_16x16x32_bf16(false, fa.v, false, fb.v, (short)0, accN, false, false);
    }

#if HAVE_ASYNC_LDS
    if (more) wait_async0();
#else
    if (more) {
      *dA[nxt] = ra;
      *dB0[nxt] = rb0;
      if (extra) *dB1[nxt] = rb1;
    }
#endif
    __syncthreads();
  }

  // epilogue: D layout -> lane holds col N=lane&15, rows M=p+8*(lane>>4)
  const int jn = n0 + wn * 16 + ml;
  const float bhr = bhh[jn], bhz = bhh[HID + jn], bhn = bhh[2 * HID + jn];
#pragma unroll
  for (int p = 0; p < 8; ++p) {
    int b = b0 + wm * 16 + p + half * 8;
    int tok = x[b * MAXLEN + t];
    if (dec_mode && t == 0) tok = 0;
    const float* gi = gitab + tok * H3;
    float r = sigf(gi[jn] + accR[p] + bhr);
    float z = sigf(gi[HID + jn] + accZ[p] + bhz);
    float n = tanh_fast(gi[2 * HID + jn] + r * (accN[p] + bhn));
    float hp = hF[(size_t)b * HID + jn];
    float hn = (1.f - z) * n + z * hp;
    hFo[(size_t)b * HID + jn] = hn;
    unsigned short hb = f2bfu(hn);
    hAo[(size_t)b * HID + jn] = hb;
    if (hall) hall[(size_t)b * HID + jn] = hb;
  }
}

// ---------------------------------------------------------------------------
// mu / logvar / z from h_enc. grid 512 (b), block 256 (l). K=1024.
// ---------------------------------------------------------------------------
__global__ void k_mulv(const float* __restrict__ henc,
                       const float* __restrict__ Wmu, const float* __restrict__ bmu,
                       const float* __restrict__ Wlv, const float* __restrict__ blv,
                       const float* __restrict__ eps,
                       float* __restrict__ mu_o, float* __restrict__ lv_o,
                       float* __restrict__ z_o) {
  __shared__ __align__(16) float hs[HID];
  int b = blockIdx.x, l = threadIdx.x;
  const float4* h4g = (const float4*)(henc + (size_t)b * HID);
  float4* h4s = (float4*)hs;
  for (int i = l; i < HID / 4; i += 256) h4s[i] = h4g[i];
  __syncthreads();
  const float4* wm4 = (const float4*)(Wmu + (size_t)l * HID);
  const float4* wl4 = (const float4*)(Wlv + (size_t)l * HID);
  float am = 0.f, al = 0.f;
  for (int k = 0; k < HID / 4; ++k) {
    float4 h = h4s[k], m = wm4[k], v = wl4[k];
    am += h.x*m.x + h.y*m.y + h.z*m.z + h.w*m.w;
    al += h.x*v.x + h.y*v.y + h.z*v.z + h.w*v.w;
  }
  float mu = am + bmu[l];
  float lv = al + blv[l];
  size_t o = (size_t)b * LAT + l;
  mu_o[o] = mu;
  lv_o[o] = lv;
  z_o[o]  = mu + eps[o] * __expf(0.5f * lv);
}

// ---------------------------------------------------------------------------
// h_dec0 = z @ W_dec^T + b_dec. grid (512, 4), block 256. K=256.
// ---------------------------------------------------------------------------
__global__ void k_dec0(const float* __restrict__ z, const float* __restrict__ Wdec,
                       const float* __restrict__ bdec,
                       float* __restrict__ hF, unsigned short* __restrict__ hA) {
  __shared__ __align__(16) float zs[LAT];
  int b = blockIdx.x;
  int h = blockIdx.y * 256 + threadIdx.x;
  const float4* z4g = (const float4*)(z + (size_t)b * LAT);
  float4* z4s = (float4*)zs;
  if (threadIdx.x < LAT / 4) z4s[threadIdx.x] = z4g[threadIdx.x];
  __syncthreads();
  const float4* w4 = (const float4*)(Wdec + (size_t)h * LAT);
  float acc = 0.f;
  for (int k = 0; k < LAT / 4; ++k) {
    float4 a = z4s[k], w = w4[k];
    acc += a.x*w.x + a.y*w.y + a.z*w.z + a.w*w.w;
  }
  acc += bdec[h];
  hF[(size_t)b * HID + h] = acc;
  hA[(size_t)b * HID + h] = f2bfu(acc);
}

// ---------------------------------------------------------------------------
// logits = h_all @ W_out^T + b_out.  A = [T*B, H] bf16, B = [H, 64] (W_out^T).
// Block: 128 rows x 64 cols; 8 waves, each wave 16 rows x 4 N-subtiles. grid 400.
// ---------------------------------------------------------------------------
__global__ void k_logits(const unsigned short* __restrict__ hall,
                         const unsigned short* __restrict__ Wout,
                         const float* __restrict__ bout, float* __restrict__ out) {
  __shared__ __align__(16) unsigned short Ah[128][40];   // [row][k]
  __shared__ __align__(16) unsigned short Bw[64][40];    // [n][k]
  const int tid = threadIdx.x, lane = tid & 31, wave = tid >> 5;
  const int half = lane >> 4, ml = lane & 15;
  const int m0 = blockIdx.x * 128;
  v8f zf = {0,0,0,0,0,0,0,0};
  v8f acc[4] = {zf, zf, zf, zf};

  for (int k0 = 0; k0 < HID; k0 += 32) {
    // A: 128 rows x 4 uint4 = 512 -> exactly 2 per thread, straight-line
    {
      int row = tid >> 2, chunk = tid & 3;
      uint4 v0 = *(const uint4*)(hall + (size_t)(m0 + row) * HID + k0 + chunk * 8);
      uint4 v1 = *(const uint4*)(hall + (size_t)(m0 + 64 + row) * HID + k0 + chunk * 8);
      int n = tid >> 2;  // B: 64 rows x 4 uint4 = 256 -> 1 per thread
      uint4 vb = *(const uint4*)(Wout + (size_t)n * HID + k0 + chunk * 8);
      *(uint4*)&Ah[row][chunk * 8]      = v0;
      *(uint4*)&Ah[64 + row][chunk * 8] = v1;
      *(uint4*)&Bw[n][chunk * 8]        = vb;
    }
    __syncthreads();
    Frag fa;
    {
      const unsigned short* ar = &Ah[wave * 16 + ml][0];
      fa.q[0] = *(const uint4*)(ar + half * 8);
      fa.q[1] = *(const uint4*)(ar + 16 + half * 8);
    }
#pragma unroll
    for (int nt = 0; nt < 4; ++nt) {
      Frag fb;
      const unsigned short* br = &Bw[nt * 16 + ml][0];
      fb.q[0] = *(const uint4*)(br + half * 8);
      fb.q[1] = *(const uint4*)(br + 16 + half * 8);
      acc[nt] = __builtin_amdgcn_wmma_f32_16x16x32_bf16(false, fa.v, false, fb.v, (short)0, acc[nt], false, false);
    }
    __syncthreads();
  }

#pragma unroll
  for (int nt = 0; nt < 4; ++nt) {
    int v = nt * 16 + ml;
    float bo = bout[v];
#pragma unroll
    for (int p = 0; p < 8; ++p) {
      int mrow = m0 + wave * 16 + p + half * 8;  // row over [T*B]
      int t = mrow / BATCH;
      int b = mrow - t * BATCH;
      out[(size_t)b * MAXLEN * VOCAB + (size_t)t * VOCAB + v] = acc[nt][p] + bo;
    }
  }
}

// ---------------------------------------------------------------------------
extern "C" void kernel_launch(void* const* d_in, const int* in_sizes, int n_in,
                              void* d_out, int out_size, void* d_ws, size_t ws_size,
                              hipStream_t stream) {
  const int*   x       = (const int*)  d_in[0];
  const float* eps     = (const float*)d_in[1];
  const float* emb     = (const float*)d_in[2];
  const float* eWih    = (const float*)d_in[3];
  const float* eWhh    = (const float*)d_in[4];
  const float* ebih    = (const float*)d_in[5];
  const float* ebhh    = (const float*)d_in[6];
  const float* Wmu     = (const float*)d_in[7];
  const float* bmu     = (const float*)d_in[8];
  const float* Wlv     = (const float*)d_in[9];
  const float* blv     = (const float*)d_in[10];
  const float* Wdec    = (const float*)d_in[11];
  const float* bdec    = (const float*)d_in[12];
  const float* dWih    = (const float*)d_in[13];
  const float* dWhh    = (const float*)d_in[14];
  const float* dbih    = (const float*)d_in[15];
  const float* dbhh    = (const float*)d_in[16];
  const float* Wout    = (const float*)d_in[17];
  const float* bout    = (const float*)d_in[18];
  float* out = (float*)d_out;

  // workspace carve-up
  char* ws = (char*)d_ws;
  size_t o = 0;
  auto take = [&](size_t bytes) { char* p = ws + o; o += (bytes + 255) & ~(size_t)255; return p; };
  float*          enc_tab = (float*)         take((size_t)VOCAB * H3 * 4);
  float*          dec_tab = (float*)         take((size_t)VOCAB * H3 * 4);
  unsigned short* eWhh_b  = (unsigned short*)take((size_t)H3 * HID * 2);
  unsigned short* dWhh_b  = (unsigned short*)take((size_t)H3 * HID * 2);
  unsigned short* Wout_b  = (unsigned short*)take((size_t)VOCAB * HID * 2);
  float*          hF[2];  hF[0] = (float*)take((size_t)BATCH * HID * 4);
                          hF[1] = (float*)take((size_t)BATCH * HID * 4);
  unsigned short* hA[2];  hA[0] = (unsigned short*)take((size_t)BATCH * HID * 2);
                          hA[1] = (unsigned short*)take((size_t)BATCH * HID * 2);
  float*          zbuf    = (float*)         take((size_t)BATCH * LAT * 4);
  unsigned short* hall    = (unsigned short*)take((size_t)MAXLEN * BATCH * HID * 2);
  if (o > ws_size) return;  // workspace too small — nothing safe to do

  const size_t LOGITS = (size_t)BATCH * MAXLEN * VOCAB;   // 3,276,800
  const size_t MUSZ   = (size_t)BATCH * LAT;              // 131,072

  // gate tables (kills the emb@Wih GEMMs)
  k_table<<<dim3(VOCAB, H3 / 256), 256, 0, stream>>>(emb, eWih, ebih, enc_tab);
  k_table<<<dim3(VOCAB, H3 / 256), 256, 0, stream>>>(emb, dWih, dbih, dec_tab);
  // weight conversions
  k_cvt_bf16<<<1024, 256, 0, stream>>>(eWhh, eWhh_b, H3 * HID);
  k_cvt_bf16<<<1024, 256, 0, stream>>>(dWhh, dWhh_b, H3 * HID);
  k_cvt_bf16<<<256, 256, 0, stream>>>(Wout, Wout_b, VOCAB * HID);
  // h0 = 0
  k_zero_h<<<512, 256, 0, stream>>>(hF[0], hA[0], BATCH * HID);

  // encoder scan
  int cur = 0;
  for (int t = 0; t < MAXLEN; ++t) {
    k_gru_step<<<dim3(BATCH / 64, HID / 32), 256, 0, stream>>>(
        hA[cur], hF[cur], eWhh_b, ebhh, enc_tab, x, t, 0,
        hF[cur ^ 1], hA[cur ^ 1], (unsigned short*)nullptr);
    cur ^= 1;
  }
  // cur == 0 after 100 steps: final h_enc is hF[0]

  k_mulv<<<BATCH, LAT, 0, stream>>>(hF[0], Wmu, bmu, Wlv, blv, eps,
                                    out + LOGITS, out + LOGITS + MUSZ, zbuf);
  k_dec0<<<dim3(BATCH, HID / 256), 256, 0, stream>>>(zbuf, Wdec, bdec, hF[0], hA[0]);

  // decoder scan (writes bf16 h_all for the logits GEMM)
  cur = 0;
  for (int t = 0; t < MAXLEN; ++t) {
    k_gru_step<<<dim3(BATCH / 64, HID / 32), 256, 0, stream>>>(
        hA[cur], hF[cur], dWhh_b, dbhh, dec_tab, x, t, 1,
        hF[cur ^ 1], hA[cur ^ 1], hall + (size_t)t * BATCH * HID);
    cur ^= 1;
  }

  k_logits<<<(MAXLEN * BATCH) / 128, 256, 0, stream>>>(hall, Wout_b, bout, out);
}
